// CharRNN_14070312862088
// MI455X (gfx1250) — matrix operand order repaired and checked by reading, not compiled, not run on hardware
//
#include <hip/hip_runtime.h>
#include <math.h>

// ---------------------------------------------------------------------------
// CharRNN on gfx1250 (MI455X): WMMA f16 GEMMs + persistent recurrence kernel
// + async memory->LDS staging (GLOBAL_LOAD_ASYNC_TO_LDS_B128, ASYNCcnt)
// ---------------------------------------------------------------------------

#define VOCAB  128
#define EMBED  256
#define HIDDEN 1024
#define BATCH  64
#define SEQ    1024

typedef __attribute__((ext_vector_type(16))) _Float16 v16h;
typedef __attribute__((ext_vector_type(8)))  float    v8f;
typedef __attribute__((ext_vector_type(4)))  unsigned u32x4;  // 16B chunk
typedef __attribute__((ext_vector_type(4)))  int      i32x4;  // 16B chunk (builtin arg type)

union Frag {
    u32x4 q[2];
    v16h  h;
};

__device__ __forceinline__ v8f wmma_f16(v16h a, v16h b, v8f c) {
    // D = A(16x32 f16) * B(32x16 f16) + C(16x16 f32)
    return __builtin_amdgcn_wmma_f32_16x16x32_f16(
        /*neg_a=*/false, a, /*neg_b=*/false, b,
        /*c_mod=*/(short)0, c, /*reuse_a=*/false, /*reuse_b=*/false);
}

// ---- async memory->LDS staging (CDNA5) ------------------------------------
#if __has_builtin(__builtin_amdgcn_global_load_async_to_lds_b128)
#define USE_ASYNC_LDS 1
#else
#define USE_ASYNC_LDS 0
#endif

__device__ __forceinline__ void async_wait0() {
#if __has_builtin(__builtin_amdgcn_s_wait_asynccnt)
    __builtin_amdgcn_s_wait_asynccnt(0);
#else
    asm volatile("s_wait_asynccnt 0" ::: "memory");
#endif
}

// Copy one 16-byte chunk global -> LDS (per lane).
__device__ __forceinline__ void stage_chunk(_Float16* lds_dst, const _Float16* gsrc) {
#if USE_ASYNC_LDS
    __builtin_amdgcn_global_load_async_to_lds_b128(
        (__attribute__((address_space(1))) i32x4*)gsrc,
        (__attribute__((address_space(3))) i32x4*)lds_dst,
        /*offset=*/0, /*cpol=*/0);
#else
    *(u32x4*)lds_dst = *(const u32x4*)gsrc;
#endif
}

__device__ __forceinline__ void stage_fence() {
#if USE_ASYNC_LDS
    async_wait0();
#endif
}

// ---------------------------------------------------------------------------
// Convert f32 weights to f16, optionally transposing into (N,K) row-major so
// WMMA B-fragments become two contiguous 16B loads per lane.
// ---------------------------------------------------------------------------
__global__ void k_cvt(const float* __restrict__ src, _Float16* __restrict__ dst,
                      int R, int C, int transpose) {
    int i = blockIdx.x * blockDim.x + threadIdx.x;
    int n = R * C;
    if (i >= n) return;
    float v;
    if (transpose) {            // dst is (C,R): dst[c*R + r] = src[r*C + c]
        int c = i / R, r = i - c * R;
        v = src[r * C + c];
    } else {
        v = src[i];
    }
    dst[i] = (_Float16)v;
}

// ---------------------------------------------------------------------------
// Zero the sync cell and the double-buffered h (h0 = 0).
// ---------------------------------------------------------------------------
__global__ void k_init(unsigned* __restrict__ sync, _Float16* __restrict__ hb) {
    int i = blockIdx.x * blockDim.x + threadIdx.x;
    if (i < 2) sync[i] = 0u;
    const int total = 2 * BATCH * HIDDEN;
    for (int j = i; j < total; j += gridDim.x * blockDim.x)
        hb[j] = (_Float16)0.0f;
}

// ---------------------------------------------------------------------------
// xW = gather(embed, x) @ Wxh + bh  ->  f32 workspace (B*T, H)
// Grid: (4096 M-tiles, 8 N-groups); block 256 = 8 waves, one 16x16 tile/wave.
// ---------------------------------------------------------------------------
__global__ void __launch_bounds__(256)
k_xw(const int* __restrict__ x,
     const _Float16* __restrict__ embedB,   // (VOCAB, EMBED) f16
     const _Float16* __restrict__ WxhT,     // (HIDDEN, EMBED) f16  == Wxh^T
     const float* __restrict__ bh,
     float* __restrict__ xw) {
    __shared__ _Float16 sA[16 * EMBED];     // 8 KB: 16 gathered embedding rows

    const int mt   = blockIdx.x;            // 0..4095  (M tile = 16 rows of B*T)
    const int ny   = blockIdx.y;            // 0..7
    const int tid  = threadIdx.x;
    const int wave = tid >> 5;
    const int lane = tid & 31;
    const int mrow = lane & 15;
    const int half = lane >> 4;

    // Stage A strip: 16 rows x 256 f16 = 512 x 16B chunks (32 chunks per row),
    // gathered by token, async memory->LDS.
    for (int idx = tid; idx < 512; idx += 256) {
        int row   = idx >> 5;
        int chunk = idx & 31;
        int token = x[mt * 16 + row];
        stage_chunk(sA + idx * 8, embedB + (size_t)token * EMBED + chunk * 8);
    }
    stage_fence();
    __syncthreads();

    const int nt   = ny * 8 + wave;         // 0..63
    const int ncol = nt * 16 + mrow;

    v8f acc;
    float bias = bh[ncol];
#pragma unroll
    for (int r = 0; r < 8; ++r) acc[r] = bias;

    const _Float16* Brow = WxhT + (size_t)ncol * EMBED;
#pragma unroll
    for (int kc = 0; kc < EMBED / 32; ++kc) {
        Frag a, b;
        const _Float16* ap = sA + mrow * EMBED + kc * 32 + half * 8;
        a.q[0] = *(const u32x4*)(ap);
        a.q[1] = *(const u32x4*)(ap + 16);
        const _Float16* bp = Brow + kc * 32 + half * 8;
        b.q[0] = *(const u32x4*)(bp);
        b.q[1] = *(const u32x4*)(bp + 16);
        acc = wmma_f16(a.h, b.h, acc);
    }

#pragma unroll
    for (int r = 0; r < 8; ++r) {
        int m = mt * 16 + r + half * 8;     // flat b*T+t row
        xw[(size_t)m * HIDDEN + ncol] = acc[r];
    }
}

// ---------------------------------------------------------------------------
// Persistent recurrence:  h_{t+1} = tanh(xW[:,t,:] + h_t @ Whh)
//                         logits[t] = H_seq[t] @ lin_W^T + lin_b
// 32 blocks x 288 threads (9 waves). Block bid: mt = bid&3 (batch tile),
// grp = bid>>2. Waves 0-7: h tiles (nt = grp*8+w). Wave 8: logits tile
// (mt, grp) for step t-1 (reuses the LDS h_t strip). One atomic grid
// barrier per step; h double-buffered in global f16.
// ---------------------------------------------------------------------------
__global__ void __launch_bounds__(288)
k_rnn(const float* __restrict__ xw,          // (B*T, H) f32
      const _Float16* __restrict__ WhhT,     // (H, H) f16 == Whh^T
      const _Float16* __restrict__ linWb,    // (V, H) f16 == lin_W
      const float* __restrict__ lin_b,
      _Float16* __restrict__ hb,             // 2 x (B, H) f16
      float* __restrict__ out_logits,        // (B, T, V) f32
      float* __restrict__ out_hlast,         // (B, H) f32
      unsigned* __restrict__ sync) {
    __shared__ _Float16 sH[16 * HIDDEN];     // 32 KB: 16 rows of h_t

    const int bid  = blockIdx.x;             // 0..31
    const int mt   = bid & 3;
    const int grp  = bid >> 2;               // 0..7
    const int tid  = threadIdx.x;
    const int wave = tid >> 5;               // 0..8
    const int lane = tid & 31;
    const int mrow = lane & 15;
    const int half = lane >> 4;

    unsigned* cnt = sync;
    unsigned* gen = sync + 1;
    unsigned  mygen = 0;
    const unsigned NB = gridDim.x;

    for (int t = 0; t <= SEQ; ++t) {
        // ---- stage h_t strip (rows mt*16 .. mt*16+15) into LDS, async ------
        const _Float16* hsrc =
            hb + (size_t)(t & 1) * (BATCH * HIDDEN) + (size_t)(mt * 16) * HIDDEN;
        for (int idx = tid; idx < (16 * HIDDEN) / 8; idx += 288)
            stage_chunk(sH + idx * 8, hsrc + idx * 8);
        stage_fence();
        __syncthreads();

        if (wave < 8 && t < SEQ) {
            // ---- h_{t+1} tile: rows mt*16.., cols nt*16.. ------------------
            const int nt   = grp * 8 + wave;        // 0..63
            const int ncol = nt * 16 + mrow;
            v8f acc;
#pragma unroll
            for (int r = 0; r < 8; ++r) {
                int bm = mt * 16 + r + half * 8;
                acc[r] = xw[((size_t)bm * SEQ + t) * HIDDEN + ncol];
            }
            const _Float16* Brow = WhhT + (size_t)ncol * HIDDEN;
            for (int kc = 0; kc < HIDDEN / 32; ++kc) {
                Frag a, b;
                const _Float16* ap = sH + mrow * HIDDEN + kc * 32 + half * 8;
                a.q[0] = *(const u32x4*)(ap);
                a.q[1] = *(const u32x4*)(ap + 16);
                const _Float16* bp = Brow + kc * 32 + half * 8;
                b.q[0] = *(const u32x4*)(bp);
                b.q[1] = *(const u32x4*)(bp + 16);
                if (kc + 1 < HIDDEN / 32)
                    __builtin_prefetch(bp + 32, 0, 0);   // global_prefetch_b8
                acc = wmma_f16(a.h, b.h, acc);
            }
            _Float16* hdst = hb + (size_t)((t + 1) & 1) * (BATCH * HIDDEN);
#pragma unroll
            for (int r = 0; r < 8; ++r) {
                int bm = mt * 16 + r + half * 8;
                float hv = tanhf(acc[r]);
                hdst[(size_t)bm * HIDDEN + ncol] = (_Float16)hv;
                if (t == SEQ - 1)
                    out_hlast[(size_t)bm * HIDDEN + ncol] = hv;
            }
        } else if (wave == 8 && t >= 1) {
            // ---- logits[t-1] tile: rows mt*16.., vocab cols grp*16.. -------
            const int vcol = grp * 16 + mrow;
            v8f acc;
            float bias = lin_b[vcol];
#pragma unroll
            for (int r = 0; r < 8; ++r) acc[r] = bias;
            const _Float16* Brow = linWb + (size_t)vcol * HIDDEN;
            for (int kc = 0; kc < HIDDEN / 32; ++kc) {
                Frag a, b;
                const _Float16* ap = sH + mrow * HIDDEN + kc * 32 + half * 8;
                a.q[0] = *(const u32x4*)(ap);
                a.q[1] = *(const u32x4*)(ap + 16);
                const _Float16* bp = Brow + kc * 32 + half * 8;
                b.q[0] = *(const u32x4*)(bp);
                b.q[1] = *(const u32x4*)(bp + 16);
                acc = wmma_f16(a.h, b.h, acc);
            }
#pragma unroll
            for (int r = 0; r < 8; ++r) {
                int bm = mt * 16 + r + half * 8;
                out_logits[((size_t)bm * SEQ + (t - 1)) * VOCAB + vcol] = acc[r];
            }
        }

        // ---- device-wide barrier (one per step) ----------------------------
        __syncthreads();
        if (tid == 0) {
            __threadfence();
            unsigned prev = atomicAdd(cnt, 1u);
            if (prev == NB - 1u) {
                *(volatile unsigned*)cnt = 0u;
                __threadfence();
                atomicAdd(gen, 1u);
            } else {
                while (*(volatile unsigned*)gen == mygen)
                    __builtin_amdgcn_s_sleep(1);
            }
            mygen++;
        }
        __syncthreads();
        __threadfence();
    }
}

// ---------------------------------------------------------------------------
// Host launcher
// ---------------------------------------------------------------------------
extern "C" void kernel_launch(void* const* d_in, const int* in_sizes, int n_in,
                              void* d_out, int out_size, void* d_ws, size_t ws_size,
                              hipStream_t stream) {
    (void)in_sizes; (void)n_in; (void)out_size; (void)ws_size;

    const int*   x       = (const int*)  d_in[0];   // (B,T)
    const float* embed_W = (const float*)d_in[1];   // (V,E)
    const float* Wxh     = (const float*)d_in[2];   // (E,H)
    const float* Whh     = (const float*)d_in[3];   // (H,H)
    const float* bh      = (const float*)d_in[4];   // (H,)
    const float* lin_W   = (const float*)d_in[5];   // (V,H)
    const float* lin_b   = (const float*)d_in[6];   // (V,)

    float* out_logits = (float*)d_out;                                // (B,T,V)
    float* out_hlast  = out_logits + (size_t)BATCH * SEQ * VOCAB;     // (B,H)

    // ---- workspace layout (256B aligned chunks) ----------------------------
    char*  ws  = (char*)d_ws;
    size_t off = 0;
    auto take = [&](size_t bytes) -> char* {
        char* p = ws + off;
        off = (off + bytes + 255) & ~(size_t)255;
        return p;
    };
    unsigned*  syncP  = (unsigned*)take(256);
    _Float16*  embedB = (_Float16*)take((size_t)VOCAB  * EMBED  * 2);
    _Float16*  WxhT   = (_Float16*)take((size_t)HIDDEN * EMBED  * 2);
    _Float16*  WhhT   = (_Float16*)take((size_t)HIDDEN * HIDDEN * 2);
    _Float16*  linWb  = (_Float16*)take((size_t)VOCAB  * HIDDEN * 2);
    _Float16*  hbP    = (_Float16*)take((size_t)2 * BATCH * HIDDEN * 2);
    float*     xwP    = (float*)   take((size_t)BATCH * SEQ * HIDDEN * 4);

    // ---- 0. init sync + h0 = 0 --------------------------------------------
    k_init<<<64, 256, 0, stream>>>(syncP, hbP);

    // ---- 1. weight conversion / transposition to f16 -----------------------
    k_cvt<<<(VOCAB * EMBED + 255) / 256, 256, 0, stream>>>(
        embed_W, embedB, VOCAB, EMBED, 0);
    k_cvt<<<(EMBED * HIDDEN + 255) / 256, 256, 0, stream>>>(
        Wxh, WxhT, EMBED, HIDDEN, 1);               // -> (H,E)
    k_cvt<<<(HIDDEN * HIDDEN + 255) / 256, 256, 0, stream>>>(
        Whh, WhhT, HIDDEN, HIDDEN, 1);              // -> (H,H) transposed
    k_cvt<<<(VOCAB * HIDDEN + 255) / 256, 256, 0, stream>>>(
        lin_W, linWb, VOCAB, HIDDEN, 0);            // already (N,K)

    // ---- 2. xW = embed[x] @ Wxh + bh ---------------------------------------
    k_xw<<<dim3((BATCH * SEQ) / 16, 8), 256, 0, stream>>>(
        x, embedB, WxhT, bh, xwP);

    // ---- 3. recurrence + logits (persistent, grid-synced) ------------------
    k_rnn<<<32, 288, 0, stream>>>(
        xwP, WhhT, linWb, lin_b, hbP, out_logits, out_hlast, syncP);
}
